// GAT_26895085208059
// MI455X (gfx1250) — matrix operand order repaired and checked by reading, not compiled, not run on hardware
//
#include <hip/hip_runtime.h>
#include <hip/hip_bf16.h>

// ---------------------------------------------------------------------------
// MI455X (gfx1250) GAT+MLP. Dense layers on v_wmma_f32_16x16x32_f16 (wave32).
// All f16 operands are padded (K->mult 32, weight N->mult 64, zero fill) so
// the GEMM inner loop is branch-free; LDS tiles are stored in WMMA fragment
// order so fragment loads are contiguous ds_load_b128.
// ---------------------------------------------------------------------------

typedef __attribute__((ext_vector_type(16))) _Float16 v16h;
typedef __attribute__((ext_vector_type(2)))  _Float16 h2;
typedef __attribute__((ext_vector_type(8)))  float    v8f;

#define TILE_M 64
#define TILE_N 64
#define TILE_K 32

// ------------------------------ WMMA GEMM ----------------------------------
// C[M,N] = A[M,Kp](lda) * B[Kp,Np](ldb) (+bias)(+relu).
// Preconditions: M % 64 == 0, Kp % 32 == 0, lda % 32 == 0, ldb % 64 == 0,
// B fully padded with zeros. Grid: (M/64, ldb/64). 128 threads = 4 waves,
// each wave owns a 32x32 quadrant = 2x2 WMMA tiles.
__global__ __launch_bounds__(128) void wmma_gemm_kernel(
    const _Float16* __restrict__ A, const _Float16* __restrict__ B,
    const float* __restrict__ bias, float* __restrict__ outF,
    _Float16* __restrict__ outH,
    int M, int N, int Kp, int lda, int ldb, int ldh, int relu)
{
    // LDS tiles stored directly in WMMA fragment order:
    // AL[mtile][lane][j] / BL[ntile][lane][j] -> fragment load = 32B contig.
    __shared__ __align__(32) _Float16 AL[4][32][16];
    __shared__ __align__(32) _Float16 BL[4][32][16];

    const int tid  = threadIdx.x;
    const int lane = tid & 31;
    const int wave = tid >> 5;
    const int l16  = lane & 15;
    const int half_sel = lane >> 4;

    const int blockM = blockIdx.x * TILE_M;
    const int blockN = blockIdx.y * TILE_N;
    const int waveM  = (wave >> 1) * 32;   // 0 or 32
    const int waveN  = (wave & 1) * 32;    // 0 or 32

    const v8f vzero = {0.f, 0.f, 0.f, 0.f, 0.f, 0.f, 0.f, 0.f};
    v8f acc[2][2] = {{vzero, vzero}, {vzero, vzero}};

    // cooperative-load roles (fixed per thread)
    const int arow = tid >> 1;             // 0..63   (A row within tile)
    const int acb  = (tid & 1) * 16;       // 0/16    (A k offset)
    const int brow = tid & 31;             // 0..31   (B k within tile)
    const int bcb  = (tid >> 5) * 16;      // 0..48   (B n offset)

    const _Float16* aptr = A + (size_t)(blockM + arow) * lda + acb;
    const _Float16* bptr = B + (size_t)brow * ldb + blockN + bcb;

    for (int k0 = 0; k0 < Kp; k0 += TILE_K) {
        if (k0 + TILE_K < Kp) {  // speculative prefetch of next K tile
            __builtin_prefetch(aptr + k0 + TILE_K, 0, 1);
            __builtin_prefetch(bptr + (size_t)(k0 + TILE_K) * ldb, 0, 1);
        }
        // ---- A: 64x32 halves, 2x b128 per thread, scatter to fragment slots
        {
            const v16h av = *(const v16h*)(aptr + k0);
            #pragma unroll
            for (int jj = 0; jj < 16; ++jj) {
                const int k = acb + jj;
                const int h = k >> 4;                             // lane group
                const int i = (k & 15) >> 1;
                AL[arow >> 4][h * 16 + (arow & 15)][2 * i + (k & 1)] = av[jj];
            }
        }
        // ---- B: 32x64 halves, 2x b128 per thread, scatter to fragment slots
        {
            const v16h bv = *(const v16h*)(bptr + (size_t)k0 * ldb);
            #pragma unroll
            for (int jj = 0; jj < 16; ++jj) {
                BL[bcb >> 4][(brow >> 4) * 16 + jj][brow & 15] = bv[jj];
            }
        }
        __syncthreads();

        // ---- fragment loads are contiguous 32B -> ds_load_b128 x2 each
        v16h bfrag[2];
        #pragma unroll
        for (int tc = 0; tc < 2; ++tc)
            bfrag[tc] = *(const v16h*)(&BL[(waveN >> 4) + tc][lane][0]);
        #pragma unroll
        for (int tr = 0; tr < 2; ++tr) {
            const v16h afrag = *(const v16h*)(&AL[(waveM >> 4) + tr][lane][0]);
            #pragma unroll
            for (int tc = 0; tc < 2; ++tc) {
                acc[tr][tc] = __builtin_amdgcn_wmma_f32_16x16x32_f16(
                    false, afrag, false, bfrag[tc], (short)0, acc[tr][tc],
                    false, false);
            }
        }
        __syncthreads();
    }

    // Epilogue. C/D layout: VGPR r -> M = r + 8*half_sel, N = lane&15.
    #pragma unroll
    for (int tr = 0; tr < 2; ++tr) {
        #pragma unroll
        for (int tc = 0; tc < 2; ++tc) {
            const int n = blockN + waveN + tc * 16 + l16;
            #pragma unroll
            for (int r = 0; r < 8; ++r) {
                const int m = blockM + waveM + tr * 16 + r + 8 * half_sel;
                float v = acc[tr][tc][r];
                if (bias && n < N) v += bias[n];
                if (relu) v = v > 0.f ? v : 0.f;
                if (outF && n < N) outF[(size_t)m * N + n] = v;
                if (outH && n < ldh)
                    outH[(size_t)m * ldh + n] = (_Float16)(n < N ? v : 0.f);
            }
        }
    }
}

// --------------------------- elementwise helpers ---------------------------
// padded convert: dst[Rp x Cp] f16 <- src[R x C] f32, zero outside.
__global__ void pad_cvt_kernel(const float* __restrict__ s,
                               _Float16* __restrict__ d,
                               int R, int C, int Cp, long long n) {
    const long long i = (long long)blockIdx.x * blockDim.x + threadIdx.x;
    if (i >= n) return;
    const int r = (int)(i / Cp), c = (int)(i % Cp);
    float v = 0.f;
    if (r < R && c < C) v = s[(size_t)r * C + c];
    d[i] = (_Float16)v;
}

__global__ void fill_u32_kernel(unsigned* __restrict__ p, unsigned v, long long n) {
    const long long i = (long long)blockIdx.x * blockDim.x + threadIdx.x;
    if (i < n) p[i] = v;
}

// order-preserving float <-> uint map (for atomic segment-max)
__device__ __forceinline__ unsigned fenc(float x) {
    unsigned u = __float_as_uint(x);
    return (u & 0x80000000u) ? ~u : (u | 0x80000000u);
}
__device__ __forceinline__ float fdec(unsigned u) {
    u = (u & 0x80000000u) ? (u & 0x7fffffffu) : ~u;
    return __uint_as_float(u);
}

// ------------------------------ GAT kernels --------------------------------
__global__ __launch_bounds__(128) void node_alpha_kernel(
    const float* __restrict__ h, const float* __restrict__ asrc,
    const float* __restrict__ adst, float* __restrict__ os,
    float* __restrict__ od, int N, int D)
{
    const int node = blockIdx.x * 4 + (threadIdx.x >> 5);
    const int lane = threadIdx.x & 31;
    if (node >= N) return;
    const float* row = h + (size_t)node * D;
    float ss = 0.f, sd = 0.f;
    for (int j = lane; j < D; j += 32) {
        const float v = row[j];
        ss += v * asrc[j];
        sd += v * adst[j];
    }
    #pragma unroll
    for (int off = 16; off > 0; off >>= 1) {
        ss += __shfl_xor(ss, off, 32);
        sd += __shfl_xor(sd, off, 32);
    }
    if (lane == 0) { os[node] = ss; od[node] = sd; }
}

// e >= E are implicit self loops (src = dst = e - E)
__global__ void edge_max_kernel(const int* __restrict__ ei, int E, int Nn,
                                const float* __restrict__ as_,
                                const float* __restrict__ ad_,
                                float* __restrict__ elog,
                                unsigned* __restrict__ mu)
{
    const int e = blockIdx.x * blockDim.x + threadIdx.x;
    if (e >= E + Nn) return;
    int s, d;
    if (e < E) { s = ei[e]; d = ei[E + e]; } else { s = d = e - E; }
    float v = as_[s] + ad_[d];
    v = v > 0.f ? v : 0.2f * v;   // leaky_relu(0.2)
    elog[e] = v;
    atomicMax(&mu[d], fenc(v));
}

__global__ void edge_exp_kernel(const int* __restrict__ ei, int E, int Nn,
                                const float* __restrict__ elog,
                                const unsigned* __restrict__ mu,
                                float* __restrict__ ex,
                                float* __restrict__ den)
{
    const int e = blockIdx.x * blockDim.x + threadIdx.x;
    if (e >= E + Nn) return;
    const int d = (e < E) ? ei[E + e] : (e - E);
    const float x = __expf(elog[e] - fdec(mu[d]));
    ex[e] = x;
    atomicAdd(&den[d], x);
}

__global__ __launch_bounds__(256) void edge_aggregate_kernel(
    const int* __restrict__ ei, int E, int Nn, const float* __restrict__ ex,
    const float* __restrict__ den, const float* __restrict__ h,
    float* __restrict__ out, int D)
{
    const int e = blockIdx.x;
    int s, d;
    if (e < E) { s = ei[e]; d = ei[E + e]; } else { s = d = e - E; }
    const float alpha = ex[e] / (den[d] + 1e-16f);
    const float* hs = h + (size_t)s * D;
    float* od = out + (size_t)d * D;
    for (int f = threadIdx.x; f < D; f += blockDim.x)
        atomicAdd(&od[f], alpha * hs[f]);
}

// out(+bias)(+relu) in f32 (stride D) and padded f16 mirror (stride ldh)
__global__ void gat_finalize_kernel(float* __restrict__ out,
                                    const float* __restrict__ b, int N, int D,
                                    int ldh, int relu, _Float16* __restrict__ x16)
{
    const long long i = (long long)blockIdx.x * blockDim.x + threadIdx.x;
    if (i >= (long long)N * ldh) return;
    const int r = (int)(i / ldh), c = (int)(i % ldh);
    if (c < D) {
        float v = out[(size_t)r * D + c] + b[c];
        if (relu) v = v > 0.f ? v : 0.f;
        out[(size_t)r * D + c] = v;
        x16[i] = (_Float16)v;
    } else {
        x16[i] = (_Float16)0.f;
    }
}

// mean-pool 50 contiguous nodes/graph + concat CLL branch -> z16 (stride ldz)
__global__ __launch_bounds__(64) void pool_concat_kernel(
    const float* __restrict__ x, const float* __restrict__ hcll,
    _Float16* __restrict__ z16, int npg, int Dg, int Dc, int ldz)
{
    const int g = blockIdx.x;
    for (int c = threadIdx.x; c < ldz; c += blockDim.x) {
        float v = 0.f;
        if (c < Dg) {
            float s = 0.f;
            for (int n = 0; n < npg; ++n)
                s += x[((size_t)g * npg + n) * Dg + c];
            v = s / (float)npg;
        } else if (c < Dg + Dc) {
            v = hcll[(size_t)g * Dc + (c - Dg)];
        }
        z16[(size_t)g * ldz + c] = (_Float16)v;
    }
}

// ------------------------------- host side ---------------------------------
static inline int PD32(int x) { return (x + 31) & ~31; }
static inline int PD64(int x) { return (x + 63) & ~63; }

extern "C" void kernel_launch(void* const* d_in, const int* in_sizes, int n_in,
                              void* d_out, int out_size, void* d_ws, size_t ws_size,
                              hipStream_t stream)
{
    const int N_NODES = 51200, N_EDGES = 204800, N_GRAPHS = 1024, NPG = 50;
    const int GAT[4][2]  = {{78,1000},{1000,700},{700,300},{300,300}};
    const int CLL[8][2]  = {{2048,2000},{2000,2000},{2000,2000},{2000,1000},
                            {1000,1000},{1000,500},{500,300},{300,300}};
    const int COMB[3][2] = {{600,200},{200,100},{100,1}};
    const int ETOT = N_EDGES + N_NODES;

    const float* cll   = (const float*)d_in[0];
    const float* x_mol = (const float*)d_in[1];
    const int*   eidx  = (const int*)d_in[2];
    auto gatW  = [&](int i){ return (const float*)d_in[4 + 4*i]; };
    auto gatAs = [&](int i){ return (const float*)d_in[5 + 4*i]; };
    auto gatAd = [&](int i){ return (const float*)d_in[6 + 4*i]; };
    auto gatB  = [&](int i){ return (const float*)d_in[7 + 4*i]; };
    auto cllW  = [&](int i){ return (const float*)d_in[20 + 2*i]; };
    auto cllB  = [&](int i){ return (const float*)d_in[21 + 2*i]; };
    auto combW = [&](int i){ return (const float*)d_in[36 + 2*i]; };
    auto combB = [&](int i){ return (const float*)d_in[37 + 2*i]; };

    char* wp = (char*)d_ws;
    auto bump = [&](size_t bytes) -> void* {
        void* r = (void*)wp;
        wp += (bytes + 255) & ~(size_t)255;
        return r;
    };
    auto padcvt = [&](const float* s, _Float16* d, int R, int C, int Rp, int Cp) {
        const long long n = (long long)Rp * Cp;
        pad_cvt_kernel<<<(unsigned)((n + 255) / 256), 256, 0, stream>>>(s, d, R, C, Cp, n);
    };
    auto fill = [&](void* p, unsigned v, long long n) {
        fill_u32_kernel<<<(unsigned)((n + 255) / 256), 256, 0, stream>>>((unsigned*)p, v, n);
    };
    auto gemm = [&](const _Float16* A, const _Float16* B, const float* bias,
                    float* oF, _Float16* oH,
                    int M, int N, int Kp, int lda, int ldb, int ldh, int relu) {
        dim3 g(M / TILE_M, ldb / TILE_N);
        wmma_gemm_kernel<<<g, 128, 0, stream>>>(A, B, bias, oF, oH,
                                                M, N, Kp, lda, ldb, ldh, relu);
    };

    // ---- weights -> padded f16 [PD32(din) x PD64(dout)] ----
    _Float16 *gW16[4], *cW16[8], *bW16[3];
    for (int i = 0; i < 4; ++i) {
        gW16[i] = (_Float16*)bump((size_t)PD32(GAT[i][0]) * PD64(GAT[i][1]) * 2);
        padcvt(gatW(i), gW16[i], GAT[i][0], GAT[i][1], PD32(GAT[i][0]), PD64(GAT[i][1]));
    }
    for (int i = 0; i < 8; ++i) {
        cW16[i] = (_Float16*)bump((size_t)PD32(CLL[i][0]) * PD64(CLL[i][1]) * 2);
        padcvt(cllW(i), cW16[i], CLL[i][0], CLL[i][1], PD32(CLL[i][0]), PD64(CLL[i][1]));
    }
    for (int i = 0; i < 3; ++i) {
        bW16[i] = (_Float16*)bump((size_t)PD32(COMB[i][0]) * PD64(COMB[i][1]) * 2);
        padcvt(combW(i), bW16[i], COMB[i][0], COMB[i][1], PD32(COMB[i][0]), PD64(COMB[i][1]));
    }

    // ---- CLL MLP branch ----
    _Float16* cin16 = (_Float16*)bump((size_t)N_GRAPHS * 2048 * 2);
    padcvt(cll, cin16, N_GRAPHS, 2048, N_GRAPHS, 2048);
    float*    c32  = (float*)bump((size_t)N_GRAPHS * 300 * 4);
    _Float16* c16a = (_Float16*)bump((size_t)N_GRAPHS * PD32(2000) * 2);
    _Float16* c16b = (_Float16*)bump((size_t)N_GRAPHS * PD32(2000) * 2);
    const _Float16* ci = cin16;
    int cilda = 2048;
    _Float16* co = c16a;
    for (int i = 0; i < 8; ++i) {
        const int din = CLL[i][0], dout = CLL[i][1];
        const int last = (i == 7);
        gemm(ci, cW16[i], cllB(i), last ? c32 : nullptr, last ? nullptr : co,
             N_GRAPHS, dout, PD32(din), cilda, PD64(dout), PD32(dout), !last);
        ci = co; cilda = PD32(dout);
        co = (co == c16a) ? c16b : c16a;
    }
    // c32: final CLL features [1024, 300] f32

    // ---- GAT branch ----
    _Float16* x16a = (_Float16*)bump((size_t)N_NODES * PD32(1000) * 2);
    _Float16* x16b = (_Float16*)bump((size_t)N_NODES * PD32(1000) * 2);
    float* h32 = (float*)bump((size_t)N_NODES * 1000 * 4);
    float* o32 = (float*)bump((size_t)N_NODES * 1000 * 4);
    float* as_ = (float*)bump((size_t)N_NODES * 4);
    float* ad_ = (float*)bump((size_t)N_NODES * 4);
    unsigned* mu = (unsigned*)bump((size_t)N_NODES * 4);
    float* den = (float*)bump((size_t)N_NODES * 4);
    float* elog = (float*)bump((size_t)ETOT * 4);
    float* exv  = (float*)bump((size_t)ETOT * 4);

    padcvt(x_mol, x16a, N_NODES, 78, N_NODES, PD32(78));
    const _Float16* xin = x16a;
    _Float16* xout = x16b;
    for (int i = 0; i < 4; ++i) {
        const int din = GAT[i][0], dout = GAT[i][1];
        // h = x @ W (bias deferred until after aggregation, per GATConv)
        gemm(xin, gW16[i], nullptr, h32, nullptr,
             N_NODES, dout, PD32(din), PD32(din), PD64(dout), 0, 0);
        node_alpha_kernel<<<(N_NODES + 3) / 4, 128, 0, stream>>>(
            h32, gatAs(i), gatAd(i), as_, ad_, N_NODES, dout);
        fill(mu, 0x007FFFFFu, N_NODES);           // fenc(-inf)
        fill(den, 0u, N_NODES);
        edge_max_kernel<<<(ETOT + 255) / 256, 256, 0, stream>>>(
            eidx, N_EDGES, N_NODES, as_, ad_, elog, mu);
        edge_exp_kernel<<<(ETOT + 255) / 256, 256, 0, stream>>>(
            eidx, N_EDGES, N_NODES, elog, mu, exv, den);
        fill(o32, 0u, (long long)N_NODES * dout);
        edge_aggregate_kernel<<<ETOT, 256, 0, stream>>>(
            eidx, N_EDGES, N_NODES, exv, den, h32, o32, dout);
        const long long nd = (long long)N_NODES * PD32(dout);
        gat_finalize_kernel<<<(unsigned)((nd + 255) / 256), 256, 0, stream>>>(
            o32, gatB(i), N_NODES, dout, PD32(dout), (i < 3) ? 1 : 0, xout);
        const _Float16* t = xin; xin = xout; xout = (_Float16*)t;
    }
    // o32: final node features [51200, 300] f32

    // ---- pool + concat (z stride PD32(600)=608, zero padded) ----
    const int ldz = PD32(600);
    _Float16* z16 = (_Float16*)bump((size_t)N_GRAPHS * ldz * 2);
    pool_concat_kernel<<<N_GRAPHS, 64, 0, stream>>>(o32, c32, z16, NPG, 300, 300, ldz);

    // ---- combiner MLP ----
    _Float16* t16a = (_Float16*)bump((size_t)N_GRAPHS * PD32(200) * 2);
    _Float16* t16b = (_Float16*)bump((size_t)N_GRAPHS * PD32(100) * 2);
    gemm(z16,  bW16[0], combB(0), nullptr, t16a,
         N_GRAPHS, 200, PD32(600), ldz, PD64(200), PD32(200), 1);
    gemm(t16a, bW16[1], combB(1), nullptr, t16b,
         N_GRAPHS, 100, PD32(200), PD32(200), PD64(100), PD32(100), 1);
    gemm(t16b, bW16[2], combB(2), (float*)d_out, nullptr,
         N_GRAPHS, 1, PD32(100), PD32(100), PD64(1), 0, 0);
}
